// PhysicsNet_22849226014830
// MI455X (gfx1250) — compile-verified, matrix operand order checked by reading, not run on hardware
//
#include <hip/hip_runtime.h>
#include <hip/hip_bf16.h>

typedef _Float16 h8  __attribute__((ext_vector_type(8)));
typedef _Float16 h16 __attribute__((ext_vector_type(16)));
typedef float    f8  __attribute__((ext_vector_type(8)));
typedef int      v4i_t __attribute__((vector_size(16)));

#define LN_EPS 1e-5f

#if __has_builtin(__builtin_amdgcn_global_load_async_to_lds_b128) && \
    __has_builtin(__builtin_amdgcn_s_wait_asynccnt)
#define HAVE_ASYNC_LDS 1
#else
#define HAVE_ASYNC_LDS 0
#endif

// Stage 32KB of f16 (16384 halves = 2048 x 16B) from global into LDS.
// Async path: GLOBAL_LOAD_ASYNC_TO_LDS_B128 (ASYNCcnt), overlapped with compute.
static __device__ inline void stage_weights_issue(const _Float16* __restrict__ src,
                                                  _Float16* dst, int tid) {
#if HAVE_ASYNC_LDS
    #pragma unroll
    for (int i = 0; i < 8; ++i) {
        int t = tid + i * 256;                       // 2048 chunks of 16B
        __builtin_amdgcn_global_load_async_to_lds_b128(
            (__attribute__((address_space(1))) v4i_t*)(void*)(src + t * 8),
            (__attribute__((address_space(3))) v4i_t*)(void*)(dst + t * 8),
            0, 0);
    }
#else
    for (int t = tid; t < 2048; t += 256) {
        *(h8*)(dst + t * 8) = *(const h8*)(src + t * 8);
    }
#endif
}
static __device__ inline void stage_weights_wait() {
#if HAVE_ASYNC_LDS
    __builtin_amdgcn_s_wait_asynccnt(0);
#endif
}

// Load a 16-half WMMA fragment slice for one lane.
// CDNA5 16-bit A/B fragment: lane L (L&15 = row/col index, L>>4 selects K-half):
//   halves 0..7  = K: kbase+0..7 ; halves 8..15 = K: kbase+16..23
static __device__ inline h16 load_frag(const _Float16* __restrict__ rowp, int kb) {
    h8 lo = *(const h8*)(rowp + kb);
    h8 hi = *(const h8*)(rowp + kb + 16);
    h16 r;
#pragma unroll
    for (int i = 0; i < 8; ++i) { r[i] = lo[i]; r[i + 8] = hi[i]; }
    return r;
}

// ---------------------------------------------------------------------------
// Kernel 0: one-time weight transpose + f32->f16 convert into workspace.
//   w2t [n*128+k] = nW2[k][n]   (128x128)
//   ew1t[n*128+k] = eW1[k][n]   (first 128 rows of 129x128)
// ---------------------------------------------------------------------------
__global__ __launch_bounds__(256) void prep_weights_kernel(
    const float* __restrict__ nW2, const float* __restrict__ eW1,
    _Float16* __restrict__ w2t, _Float16* __restrict__ ew1t)
{
    int t = blockIdx.x * 256 + threadIdx.x;
    if (t < 16384) {
        int n = t >> 7, k = t & 127;
        w2t[t] = (_Float16)nW2[k * 128 + n];
    } else if (t < 32768) {
        int u = t - 16384;
        int n = u >> 7, k = u & 127;
        ew1t[u] = (_Float16)eW1[k * 128 + n];
    }
}

// ---------------------------------------------------------------------------
// Kernel 1: node MLP + LayerNorm -> latent (f16) + kinetic reduction.
// 64 nodes per 256-thread block (8 wave32).
// ---------------------------------------------------------------------------
__global__ __launch_bounds__(256) void node_mlp_kernel(
    const float* __restrict__ x,          // (N,6,2)
    const float* __restrict__ node_mass,  // (N,1)
    const float* __restrict__ nW1,        // (3,128)
    const float* __restrict__ nb1,        // (128)
    const _Float16* __restrict__ w2t,     // (128,128) f16, pre-transposed
    const float* __restrict__ nb2,        // (128)
    const float* __restrict__ ng,         // (128)
    const float* __restrict__ nbeta,      // (128)
    _Float16* __restrict__ latent,        // (N,128) f16
    float* __restrict__ out,              // out[0] += kinetic
    int N)
{
    __shared__ __align__(16) _Float16 sA[64 * 128];    // h1 then h2 (f16)
    __shared__ __align__(16) _Float16 sB[128 * 128];   // W2^T: sB[n*128+k]
    __shared__ float sPos[64 * 3];
    __shared__ float sMu[64];
    __shared__ float sRs[64];
    __shared__ float sK[64];

    const int tid  = threadIdx.x;
    const int base = blockIdx.x * 64;

    // Kick off async W2^T staging immediately (overlaps everything below)
    stage_weights_issue(w2t, sB, tid);

    // Stage positions (x[:,0:3,-1]); fold kinetic term (x[:,3:6,-1], mass)
    if (tid < 64) {
        int n = base + tid;
        float p0 = 0.f, p1 = 0.f, p2 = 0.f, kin = 0.f;
        if (n < N) {
            p0 = x[n * 12 + 0 * 2 + 1];
            p1 = x[n * 12 + 1 * 2 + 1];
            p2 = x[n * 12 + 2 * 2 + 1];
            float v0 = x[n * 12 + 3 * 2 + 1];
            float v1 = x[n * 12 + 4 * 2 + 1];
            float v2 = x[n * 12 + 5 * 2 + 1];
            kin = 0.5f * node_mass[n] * (v0 * v0 + v1 * v1 + v2 * v2);
        }
        sPos[tid * 3 + 0] = p0; sPos[tid * 3 + 1] = p1; sPos[tid * 3 + 2] = p2;
        sK[tid] = kin;
    }
    __syncthreads();

    // h1 = relu(pos @ W1 + b1) -> sA (f16)
    for (int t = tid; t < 64 * 128; t += 256) {
        int r = t >> 7, h = t & 127;
        float acc = nb1[h]
                  + sPos[r * 3 + 0] * nW1[0 * 128 + h]
                  + sPos[r * 3 + 1] * nW1[1 * 128 + h]
                  + sPos[r * 3 + 2] * nW1[2 * 128 + h];
        sA[t] = (_Float16)fmaxf(acc, 0.f);
    }
    stage_weights_wait();     // our wave's async LDS writes done
    __syncthreads();          // everyone's done -> sA/sB consistent

    // h2 = h1 @ W2 via WMMA f32_16x16x32_f16.
    // wave w -> node-group g = w&3 (16 rows), column half nh = w>>2 (64 cols).
    const int wave = tid >> 5, lane = tid & 31;
    const int g = wave & 3, nh = wave >> 2;
    const int lr  = lane & 15;
    const int kbl = (lane >> 4) * 8;
    const _Float16* arow = sA + (g * 16 + lr) * 128;

    f8 acc[4] = {};
#pragma unroll
    for (int kt = 0; kt < 4; ++kt) {
        h16 a = load_frag(arow, kt * 32 + kbl);
#pragma unroll
        for (int nt = 0; nt < 4; ++nt) {
            int n = nh * 64 + nt * 16 + lr;
            h16 b = load_frag(sB + n * 128, kt * 32 + kbl);
            acc[nt] = __builtin_amdgcn_wmma_f32_16x16x32_f16(
                false, a, false, b, (short)0, acc[nt], false, false);
        }
    }
    __syncthreads();   // all h1 reads done before overwriting sA

    // D layout: VGPR r holds M = r + (lane>>4)*8, N = lane&15 within tile.
#pragma unroll
    for (int nt = 0; nt < 4; ++nt) {
        int col = nh * 64 + nt * 16 + lr;
        float bias = nb2[col];
#pragma unroll
        for (int r = 0; r < 8; ++r) {
            int m = g * 16 + r + (lane >> 4) * 8;
            sA[m * 128 + col] = (_Float16)(acc[nt][r] + bias);
        }
    }
    __syncthreads();

    // Row stats for LayerNorm
    if (tid < 64) {
        float s = 0.f;
        for (int c = 0; c < 128; ++c) s += (float)sA[tid * 128 + c];
        float mu = s * (1.f / 128.f);
        float v = 0.f;
        for (int c = 0; c < 128; ++c) {
            float d = (float)sA[tid * 128 + c] - mu;
            v += d * d;
        }
        sMu[tid] = mu;
        sRs[tid] = rsqrtf(v * (1.f / 128.f) + LN_EPS);
    }
    __syncthreads();

    // Normalize, scale/shift, coalesced f16 store
    for (int t = tid; t < 64 * 128; t += 256) {
        int r = t >> 7, c = t & 127;
        int n = base + r;
        if (n < N) {
            float o = ((float)sA[t] - sMu[r]) * sRs[r] * ng[c] + nbeta[c];
            latent[(size_t)n * 128 + c] = (_Float16)o;
        }
    }

    // Kinetic: one serial 64-sum per block (negligible), one atomic per block
    if (tid == 0) {
        float s = 0.f;
        for (int i = 0; i < 64; ++i) s += sK[i];
        atomicAdd(&out[0], s);
    }
}

// ---------------------------------------------------------------------------
// Kernel 2: element gather + MLP + size-1 LN + sum, 64 elems per block
// ---------------------------------------------------------------------------
__global__ __launch_bounds__(256) void elem_kernel(
    const _Float16* __restrict__ latent,  // (N,128) f16
    const float* __restrict__ mats,       // (E,1)
    const int*   __restrict__ e2n,        // (E,8)
    const _Float16* __restrict__ ew1t,    // (128,128) f16: eW1[0:128]^T
    const float* __restrict__ eW1,        // (129,128) f32 (for material row)
    const float* __restrict__ eb1,        // (128)
    const float* __restrict__ eW2,        // (128,1)
    const float* __restrict__ eb2,        // (1)
    const float* __restrict__ eg,         // (1)
    const float* __restrict__ ebeta,      // (1)
    float* __restrict__ out, int E)
{
    __shared__ __align__(16) _Float16 sF[64 * 128];    // gathered feats, then h1
    __shared__ __align__(16) _Float16 sW[128 * 128];   // eW1[0:128]^T
    __shared__ float sMat[64];
    __shared__ int   sIdx[64 * 8];
    __shared__ float sRed[256];

    const int tid  = threadIdx.x;
    const int base = blockIdx.x * 64;

    // Async weight staging overlaps the (long-latency) gather below
    stage_weights_issue(ew1t, sW, tid);

    for (int t = tid; t < 64 * 8; t += 256) {
        int e = base + (t >> 3);
        sIdx[t] = (e < E) ? e2n[(size_t)e * 8 + (t & 7)] : 0;
    }
    if (tid < 64) {
        int e = base + tid;
        sMat[tid] = (e < E) ? mats[e] : 0.f;
    }
    __syncthreads();

    // Gather-sum 8 latent rows per element (f16 traffic, coalesced in c)
    for (int t = tid; t < 64 * 128; t += 256) {
        int r = t >> 7, c = t & 127;
        int e = base + r;
        float s = 0.f;
#pragma unroll
        for (int j = 0; j < 8; ++j)
            s += (float)latent[(size_t)sIdx[r * 8 + j] * 128 + c];
        sF[t] = (e < E) ? (_Float16)s : (_Float16)0.f;
    }
    stage_weights_wait();
    __syncthreads();

    // h1(pre-act) = feat @ eW1[0:128] via WMMA; material row + bias in epilogue
    const int wave = tid >> 5, lane = tid & 31;
    const int g = wave & 3, nh = wave >> 2;
    const int lr  = lane & 15;
    const int kbl = (lane >> 4) * 8;
    const _Float16* arow = sF + (g * 16 + lr) * 128;

    f8 acc[4] = {};
#pragma unroll
    for (int kt = 0; kt < 4; ++kt) {
        h16 a = load_frag(arow, kt * 32 + kbl);
#pragma unroll
        for (int nt = 0; nt < 4; ++nt) {
            int n = nh * 64 + nt * 16 + lr;
            h16 b = load_frag(sW + n * 128, kt * 32 + kbl);
            acc[nt] = __builtin_amdgcn_wmma_f32_16x16x32_f16(
                false, a, false, b, (short)0, acc[nt], false, false);
        }
    }
    __syncthreads();

#pragma unroll
    for (int nt = 0; nt < 4; ++nt) {
        int col = nh * 64 + nt * 16 + lr;
        float w128 = eW1[128 * 128 + col];   // material row of eW1 (row 128)
        float b    = eb1[col];
#pragma unroll
        for (int r = 0; r < 8; ++r) {
            int m = g * 16 + r + (lane >> 4) * 8;
            float hv = acc[nt][r] + sMat[m] * w128 + b;
            sF[m * 128 + col] = (_Float16)fmaxf(hv, 0.f);
        }
    }
    __syncthreads();

    // h = h1 @ eW2 + eb2 ; LayerNorm over last dim of size 1 (faithful)
    float val = 0.f;
    if (tid < 64) {
        int e = base + tid;
        float h = eb2[0];
        for (int c = 0; c < 128; ++c)
            h += (float)sF[tid * 128 + c] * eW2[c];
        float mu  = h;
        float var = 0.f;
        float pe  = (h - mu) * rsqrtf(var + LN_EPS) * eg[0] + ebeta[0];
        val = (e < E) ? pe : 0.f;
    }
    sRed[tid] = val;
    __syncthreads();
    for (int s = 128; s > 0; s >>= 1) {
        if (tid < s) sRed[tid] += sRed[tid + s];
        __syncthreads();
    }
    if (tid == 0) atomicAdd(&out[1], sRed[0]);
}

// ---------------------------------------------------------------------------
extern "C" void kernel_launch(void* const* d_in, const int* in_sizes, int n_in,
                              void* d_out, int out_size, void* d_ws, size_t ws_size,
                              hipStream_t stream) {
    const float* x     = (const float*)d_in[0];
    const float* mass  = (const float*)d_in[1];
    const float* mats  = (const float*)d_in[2];
    const int*   e2n   = (const int*)  d_in[3];
    const float* nW1   = (const float*)d_in[4];
    const float* nb1   = (const float*)d_in[5];
    const float* nW2   = (const float*)d_in[6];
    const float* nb2   = (const float*)d_in[7];
    const float* ng    = (const float*)d_in[8];
    const float* nbeta = (const float*)d_in[9];
    const float* eW1   = (const float*)d_in[10];
    const float* eb1   = (const float*)d_in[11];
    const float* eW2   = (const float*)d_in[12];
    const float* eb2   = (const float*)d_in[13];
    const float* eg    = (const float*)d_in[14];
    const float* ebeta = (const float*)d_in[15];

    const int N = in_sizes[1];          // node_mass has N elements
    const int E = in_sizes[2];          // element_materials has E elements

    float* out = (float*)d_out;

    // Workspace layout: latent (N*128 f16), then W2^T f16, then eW1^T f16
    char* ws = (char*)d_ws;
    _Float16* latent = (_Float16*)ws;
    size_t latent_bytes = ((size_t)N * 128 * sizeof(_Float16) + 255) & ~(size_t)255;
    _Float16* w2t  = (_Float16*)(ws + latent_bytes);
    _Float16* ew1t = w2t + 16384;

    (void)hipMemsetAsync(d_out, 0, (size_t)out_size * sizeof(float), stream);

    prep_weights_kernel<<<128, 256, 0, stream>>>(nW2, eW1, w2t, ew1t);
    node_mlp_kernel<<<(N + 63) / 64, 256, 0, stream>>>(
        x, mass, nW1, nb1, w2t, nb2, ng, nbeta, latent, out, N);
    elem_kernel<<<(E + 63) / 64, 256, 0, stream>>>(
        latent, mats, e2n, ew1t, eW1, eb1, eW2, eb2, eg, ebeta, out, E);
}
